// SelfAttention_33079838114674
// MI455X (gfx1250) — matrix operand order repaired and checked by reading, not compiled, not run on hardware
//
#include <hip/hip_runtime.h>
#include <hip/hip_bf16.h>
#include <math.h>

#define DIM    512
#define SEQ    2048
#define NBATCH 32
#define MTOT   (NBATCH * SEQ)

typedef __attribute__((ext_vector_type(16))) _Float16 v16h;
typedef __attribute__((ext_vector_type(8)))  _Float16 v8h;
typedef __attribute__((ext_vector_type(8)))  float    v8f;

// Build a 16-element f16 A-fragment from two contiguous 8-element runs
// (contiguous K order, as for B: per lane K = ka..ka+15, ka = 16 * (lane / 16)).
static __device__ __forceinline__ v16h pack16(v8h lo, v8h hi) {
    v16h a;
#pragma unroll
    for (int i = 0; i < 8; ++i) { a[i] = lo[i]; a[8 + i] = hi[i]; }
    return a;
}

// ---------------------------------------------------------------------------
// One-time f32 -> f16 conversion (memory bound: ~8us at 23.3 TB/s for Ez).
// use_z selects layer z of a [6, DIM, DIM] weight tensor on-device.
// ---------------------------------------------------------------------------
__global__ __launch_bounds__(256) void cvt_f32_to_f16(
    const float* __restrict__ src, const int* __restrict__ zp, int use_z,
    _Float16* __restrict__ dst)
{
    const float* __restrict__ s =
        src + (use_z ? (size_t)(*zp) * DIM * DIM : (size_t)0);
    const size_t i = ((size_t)blockIdx.x * blockDim.x + threadIdx.x) * 8;
    const float4 a = *(const float4*)(s + i);
    const float4 b = *(const float4*)(s + i + 4);
    v8h o;
    o[0] = (_Float16)a.x; o[1] = (_Float16)a.y;
    o[2] = (_Float16)a.z; o[3] = (_Float16)a.w;
    o[4] = (_Float16)b.x; o[5] = (_Float16)b.y;
    o[6] = (_Float16)b.z; o[7] = (_Float16)b.w;
    *(v8h*)(dst + i) = o;
}

// ---------------------------------------------------------------------------
// Projection GEMM (pure f16): out = Xh @ Wh^T.
// Wave tile = 16x64 (4 independent accumulator chains, 4x A reuse).
// Software-pipelined: fragments for iteration d+32 are loaded before the
// WMMAs of iteration d, so waits need not drain loadcnt to 0.
// transpose_out == 0 : out f16 [MTOT][DIM]   (Q, K)
// transpose_out == 1 : out f16 [B][DIM][SEQ] (V transposed, for P*V)
// ---------------------------------------------------------------------------
__global__ __launch_bounds__(128) void proj_kernel(
    const _Float16* __restrict__ Xh, const _Float16* __restrict__ Wh,
    _Float16* __restrict__ out, int transpose_out)
{
    const int lane = threadIdx.x & 31;
    const int wave = threadIdx.x >> 5;
    const int rc   = lane & 15;            // A-row / B-col within a 16x16 tile
    const int ka   = (lane >> 4) * 16;     // A-fragment K base (contig 16 run)
    const int kb   = (lane >> 4) * 16;     // B-fragment K base (contig 16 run)
    const int moff = (lane >> 4) * 8;
    const int mbase  = blockIdx.y * 16;
    const int ebase0 = blockIdx.x * 256 + wave * 64;

    const _Float16* __restrict__ xrow = Xh + (size_t)(mbase + rc) * DIM;
    const _Float16* __restrict__ wrow = Wh + (size_t)(ebase0 + rc) * DIM;

    v8f c[4] = {};

    // pipeline prologue: fragments for d = 0
    v16h a_c = pack16(*(const v8h*)(xrow + ka), *(const v8h*)(xrow + ka + 8));
    v16h b_c[4];
#pragma unroll
    for (int j = 0; j < 4; ++j)
        b_c[j] = *(const v16h*)(wrow + (size_t)(16 * j) * DIM + kb);

#pragma unroll 2
    for (int d = 0; d < DIM; d += 32) {
        const int dn = (d + 32) & (DIM - 1);   // wraps on last iter (dead load)
        const v16h a_n = pack16(*(const v8h*)(xrow + dn + ka),
                                *(const v8h*)(xrow + dn + ka + 8));
        v16h b_n[4];
#pragma unroll
        for (int j = 0; j < 4; ++j)
            b_n[j] = *(const v16h*)(wrow + (size_t)(16 * j) * DIM + dn + kb);
#pragma unroll
        for (int j = 0; j < 4; ++j)
            c[j] = __builtin_amdgcn_wmma_f32_16x16x32_f16(false, a_c, false, b_c[j],
                                                          (short)0, c[j],
                                                          false, false);
        a_c = a_n;
#pragma unroll
        for (int j = 0; j < 4; ++j) b_c[j] = b_n[j];
    }

    // C/D layout: VGPR r -> row M = moff + r, column N = rc
    if (transpose_out == 0) {
#pragma unroll
        for (int j = 0; j < 4; ++j)
#pragma unroll
            for (int r = 0; r < 8; ++r)
                out[(size_t)(mbase + moff + r) * DIM + (ebase0 + 16 * j + rc)] =
                    (_Float16)c[j][r];
    } else {
        const int m0 = mbase + moff;          // 16-row tiles never cross a batch
        const int bb = m0 >> 11;              // m0 / SEQ
        const int l0 = m0 & (SEQ - 1);
#pragma unroll
        for (int j = 0; j < 4; ++j) {
            v8h pk;
#pragma unroll
            for (int r = 0; r < 8; ++r) pk[r] = (_Float16)c[j][r];
            *(v8h*)(out + ((size_t)bb * DIM + (ebase0 + 16 * j + rc)) * SEQ + l0) = pk;
        }
    }
}

// ---------------------------------------------------------------------------
// Causal flash attention.
// Block = 4 waves, one (batch, 16-query tile). Key tiles of 64.
// Q tile staged once in LDS; K/V B-fragments software-pipelined from global.
//   wave w: S subtile keys [k0+16w, +16) -- two interleaved WMMA chains over D
//           O columns [128w, 128w+128)   -- 8 C fragments, 16 WMMAs / tile
// Online softmax (m, l) by one thread per query row through LDS.
// ---------------------------------------------------------------------------
__global__ __launch_bounds__(128) void attn_kernel(
    const _Float16* __restrict__ Q, const _Float16* __restrict__ K,
    const _Float16* __restrict__ Vt, float* __restrict__ out)
{
    __shared__ _Float16 sQ[16][DIM];
    __shared__ float    sS[16][64];
    __shared__ _Float16 sP[16][64];
    __shared__ float    sScale[16];
    __shared__ float    sL[16];

    const int lane = threadIdx.x & 31;
    const int wave = threadIdx.x >> 5;
    const int rc   = lane & 15;
    const int ka   = (lane >> 4) * 16;
    const int kb   = (lane >> 4) * 16;
    const int moff = (lane >> 4) * 8;
    const int qbase = blockIdx.x * 16;
    const size_t boff = (size_t)blockIdx.y * SEQ * DIM;

    const _Float16* __restrict__ Qb = Q  + boff;
    const _Float16* __restrict__ Kb = K  + boff;
    const _Float16* __restrict__ Vb = Vt + boff;   // [DIM][SEQ]

    // ---- stage Q tile (16 x 512 f16 = 16 KB) into LDS ----
#pragma unroll
    for (int i = 0; i < 8; ++i) {
        const int off = threadIdx.x * 64 + i * 8;
        const int r = off >> 9, cc = off & (DIM - 1);
        *(v8h*)&sQ[r][cc] = *(const v8h*)(Qb + (size_t)(qbase + r) * DIM + cc);
    }
    __syncthreads();

    v8f o[8] = {};
    float mrow = -INFINITY, lrow = 0.0f;           // live in threads 0..15
    const float sm = 0.04419417382415922f;         // 1/sqrt(512)

    const int ktiles = (qbase + 16 + 63) >> 6;     // keys 0 .. qbase+15
    for (int t = 0; t < ktiles; ++t) {
        const int k0 = t << 6;

        // ---- S = (Q K^T) subtile: 2 accumulator chains, pipelined K loads ---
        const _Float16* __restrict__ krow =
            Kb + (size_t)(k0 + wave * 16 + rc) * DIM;
        if (t + 1 < ktiles)
            __builtin_prefetch(Kb + (size_t)(k0 + 64 + wave * 16 + rc) * DIM, 0, 0);

        v8f s0 = {}, s1 = {};
        v16h bk0 = *(const v16h*)(krow + kb);
        v16h bk1 = *(const v16h*)(krow + 32 + kb);
#pragma unroll 2
        for (int d = 0; d < DIM; d += 64) {
            const int dn = (d + 64) & (DIM - 1);   // wraps on last iter
            const v16h bn0 = *(const v16h*)(krow + dn + kb);
            const v16h bn1 = *(const v16h*)(krow + dn + 32 + kb);
            const v16h a0 = pack16(*(const v8h*)&sQ[rc][d + ka],
                                   *(const v8h*)&sQ[rc][d + ka + 8]);
            s0 = __builtin_amdgcn_wmma_f32_16x16x32_f16(false, a0, false, bk0,
                                                        (short)0, s0, false, false);
            const v16h a1 = pack16(*(const v8h*)&sQ[rc][d + 32 + ka],
                                   *(const v8h*)&sQ[rc][d + 32 + ka + 8]);
            s1 = __builtin_amdgcn_wmma_f32_16x16x32_f16(false, a1, false, bk1,
                                                        (short)0, s1, false, false);
            bk0 = bn0; bk1 = bn1;
        }
#pragma unroll
        for (int r = 0; r < 8; ++r)
            sS[moff + r][wave * 16 + rc] = (s0[r] + s1[r]) * sm;
        __syncthreads();

        // ---- online softmax, one thread per query row ----
        if (threadIdx.x < 16) {
            const int r = threadIdx.x;
            int kmax = (qbase + r) - k0 + 1;        // causal: keys <= q
            if (kmax > 64) kmax = 64;
            float mx = -INFINITY;
            for (int j = 0; j < kmax; ++j) mx = fmaxf(mx, sS[r][j]);
            const float mnew  = fmaxf(mrow, mx);
            const float scale = __expf(mrow - mnew);
            float sum = 0.0f;
            for (int j = 0; j < 64; ++j) {
                const float p = (j < kmax) ? __expf(sS[r][j] - mnew) : 0.0f;
                sP[r][j] = (_Float16)p;
                sum += p;
            }
            lrow = lrow * scale + sum;
            mrow = mnew;
            sScale[r] = scale;
        }
        __syncthreads();

        // ---- rescale O, then O += P * V (pipelined V loads) ----
#pragma unroll
        for (int r = 0; r < 8; ++r) {
            const float f = sScale[moff + r];
#pragma unroll
            for (int j = 0; j < 8; ++j) o[j][r] *= f;
        }

        const v8h* __restrict__ prow = (const v8h*)&sP[rc][0];
        const v16h aP0 = pack16(prow[(ka) >> 3],      prow[(ka + 8) >> 3]);
        const v16h aP1 = pack16(prow[(32 + ka) >> 3], prow[(32 + ka + 8) >> 3]);

        auto vldV = [&](int idx) -> v16h {           // idx = ch*8 + j
            const int j = idx & 7, ch = idx >> 3;
            const int e = wave * 128 + j * 16 + rc;
            return *(const v16h*)(Vb + (size_t)e * SEQ + k0 + ch * 32 + kb);
        };

        v16h bv = vldV(0);
#pragma unroll
        for (int idx = 0; idx < 16; ++idx) {
            const v16h bn = vldV(idx < 15 ? idx + 1 : 15);  // last is dead
            const v16h a = (idx >> 3) ? aP1 : aP0;
            o[idx & 7] = __builtin_amdgcn_wmma_f32_16x16x32_f16(false, a, false, bv,
                                                                (short)0, o[idx & 7],
                                                                false, false);
            bv = bn;
        }
        __syncthreads();
    }

    if (threadIdx.x < 16) sL[threadIdx.x] = lrow;
    __syncthreads();

    float* __restrict__ ob = out + boff;
#pragma unroll
    for (int r = 0; r < 8; ++r) {
        const float inv = 1.0f / sL[moff + r];
        const int q = qbase + moff + r;
#pragma unroll
        for (int j = 0; j < 8; ++j)
            ob[(size_t)q * DIM + wave * 128 + j * 16 + rc] = o[j][r] * inv;
    }
}

// ---------------------------------------------------------------------------
extern "C" void kernel_launch(void* const* d_in, const int* in_sizes, int n_in,
                              void* d_out, int out_size, void* d_ws, size_t ws_size,
                              hipStream_t stream)
{
    (void)in_sizes; (void)n_in; (void)out_size; (void)ws_size;

    const float* Ez = (const float*)d_in[0];
    const float* Wq = (const float*)d_in[1];
    const float* Wk = (const float*)d_in[2];
    const float* Wv = (const float*)d_in[3];
    const int*   zp = (const int*)d_in[4];
    float* outp = (float*)d_out;

    const size_t EZN = (size_t)MTOT * DIM;          // 33.5M f16
    const size_t WN  = (size_t)DIM * DIM;           // 262K f16 per layer

    _Float16* Ezh = (_Float16*)d_ws;                // 64 MB
    _Float16* Wh  = Ezh + EZN;                      // 1.5 MB (3 slices)
    _Float16* Qh  = Wh + 3 * WN;                    // 64 MB
    _Float16* Kh  = Qh + EZN;                       // 64 MB
    _Float16* Vth = Kh + EZN;                       // 64 MB (transposed V)

    // one-time f32 -> f16 conversions
    cvt_f32_to_f16<<<EZN / (256 * 8), 256, 0, stream>>>(Ez, zp, 0, Ezh);
    cvt_f32_to_f16<<<WN  / (256 * 8), 256, 0, stream>>>(Wq, zp, 1, Wh);
    cvt_f32_to_f16<<<WN  / (256 * 8), 256, 0, stream>>>(Wk, zp, 1, Wh + WN);
    cvt_f32_to_f16<<<WN  / (256 * 8), 256, 0, stream>>>(Wv, zp, 1, Wh + 2 * WN);

    // Q/K/V projections (pure f16 WMMA GEMMs)
    dim3 pgrid(DIM / 256, MTOT / 16);
    proj_kernel<<<pgrid, 128, 0, stream>>>(Ezh, Wh,          Qh,  0);
    proj_kernel<<<pgrid, 128, 0, stream>>>(Ezh, Wh + WN,     Kh,  0);
    proj_kernel<<<pgrid, 128, 0, stream>>>(Ezh, Wh + 2 * WN, Vth, 1);

    // causal flash attention
    dim3 agrid(SEQ / 16, NBATCH);
    attn_kernel<<<agrid, 128, 0, stream>>>(Qh, Kh, Vth, outp);
}
